// MultiheadAttention_72911364817064
// MI455X (gfx1250) — matrix operand order repaired and checked by reading, not compiled, not run on hardware
//
#include <hip/hip_runtime.h>
#include <hip/hip_bf16.h>

// ---------------------------------------------------------------------------
// iRPE 'product' multi-head attention, MI455X (gfx1250), wave32 + WMMA f16.
// B=8, N=1024, C=768, H=12, HD=64, NB=49, SCALE=0.125
// d_out = [ out: B*N*C f32 | attn: B*H*N*N f32 ]
// HBM-bound (~1.5 GB traffic @ 23.3 TB/s): attn output streamed with NT
// stores so K/V/rp/tk/tq stay L2-resident; GEMM work on v_wmma f16/f32-acc.
// ---------------------------------------------------------------------------

#define BB 8
#define NN 1024
#define CC 768
#define HH 12
#define HD 64
#define NB 49
#define SCALE 0.125f

typedef __attribute__((ext_vector_type(16))) _Float16 v16h;
typedef __attribute__((ext_vector_type(8)))  _Float16 v8h;
typedef __attribute__((ext_vector_type(8)))  float    v8f;

__device__ __forceinline__ v8f wmma_f16(v16h a, v16h b, v8f c) {
  return __builtin_amdgcn_wmma_f32_16x16x32_f16(false, a, false, b,
                                                (short)0, c, false, false);
}

// A fragment, 16x32 f16, row-major source with leading dim ld (halves).
//   lanes 0-15 : M=lane, K = k0+0..7 (elems0-7), k0+16..23 (elems8-15)
//   lanes16-31 : M=lane-16, K = k0+8..15, k0+24..31
__device__ __forceinline__ v16h load_a_frag(const _Float16* base, int ld, int lane) {
  const _Float16* p = base + (size_t)(lane & 15) * (size_t)ld + ((lane & 16) ? 8 : 0);
  v8h lo = *(const v8h*)p;
  v8h hi = *(const v8h*)(p + 16);
  v16h a;
#pragma unroll
  for (int e = 0; e < 8; ++e) { a[e] = lo[e]; a[8 + e] = hi[e]; }
  return a;
}

// B fragment, 32x16 f16.  B[kk,n] = src[(n0+n)*ld + k0+kk] (K contiguous/col).
//   lanes 0-15 : col n=lane, K = k0+0..7, k0+16..23 ; lanes16-31 : col n=lane-16, K = k0+8..15, k0+24..31
__device__ __forceinline__ v16h load_b_frag(const _Float16* base, int ld, int lane) {
  const _Float16* p = base + (size_t)(lane & 15) * (size_t)ld + ((lane & 16) ? 8 : 0);
  v8h lo = *(const v8h*)p;
  v8h hi = *(const v8h*)(p + 16);
  v16h b;
#pragma unroll
  for (int e = 0; e < 8; ++e) { b[e] = lo[e]; b[8 + e] = hi[e]; }
  return b;
}

// Fire-and-forget global->LDS async copy (ASYNCcnt path, ISA 15.18.3).
__device__ __forceinline__ void async_f32_to_lds(const float* gptr, const float* lptr) {
  unsigned ldsoff = (unsigned)(uintptr_t)lptr;           // addr[31:0] = LDS offset
  unsigned long long ga = (unsigned long long)(uintptr_t)gptr;
  asm volatile("global_load_async_to_lds_b32 %0, %1, off"
               :: "v"(ldsoff), "v"(ga) : "memory");
}
__device__ __forceinline__ void wait_asynccnt0() {
  asm volatile("s_wait_asynccnt 0" ::: "memory");
}

// ---------------------------------------------------------------------------
// Kernel 1: pack q,k -> [B,H,N,HD] f16 ; v -> [B,H,HD,N] f16 (transposed)
// q/k/v are read exactly once -> stream with non-temporal loads.
// ---------------------------------------------------------------------------
__global__ void pack_qkv_kernel(const float* __restrict__ q,
                                const float* __restrict__ k,
                                const float* __restrict__ v,
                                _Float16* __restrict__ qh,
                                _Float16* __restrict__ kh,
                                _Float16* __restrict__ vt) {
  size_t idx = (size_t)blockIdx.x * blockDim.x + threadIdx.x;
  if (idx >= (size_t)BB * NN * CC) return;
  int c = (int)(idx % CC);
  size_t bn = idx / CC;
  int n = (int)(bn % NN);
  int b = (int)(bn / NN);
  int h = c / HD, d = c % HD;
  int bh = b * HH + h;
  qh[((size_t)bh * NN + n) * HD + d] = (_Float16)__builtin_nontemporal_load(q + idx);
  kh[((size_t)bh * NN + n) * HD + d] = (_Float16)__builtin_nontemporal_load(k + idx);
  vt[((size_t)bh * HD + d) * NN + n] = (_Float16)__builtin_nontemporal_load(v + idx);
}

// ---------------------------------------------------------------------------
// Kernel 2: proj_w -> f16 (row major; B[k,n]=proj_w[n,k] is K-contiguous)
// ---------------------------------------------------------------------------
__global__ void pack_w_kernel(const float* __restrict__ w, _Float16* __restrict__ w16) {
  size_t idx = (size_t)blockIdx.x * blockDim.x + threadIdx.x;
  if (idx < (size_t)CC * CC) w16[idx] = (_Float16)w[idx];
}

// ---------------------------------------------------------------------------
// Kernel 3: RPE bias tables.  row = bh*N + i
//   tk[row,u] = q_row . w_rpe_k[:,u] ;  tq[row,u] = SCALE * k_row . w_rpe_q[:,u]
// ---------------------------------------------------------------------------
__global__ void bias_kernel(const _Float16* __restrict__ qh,
                            const _Float16* __restrict__ kh,
                            const float* __restrict__ wk,   // w_rpe_k [HD,NB]
                            const float* __restrict__ wq,   // w_rpe_q [HD,NB]
                            float* __restrict__ tk,
                            float* __restrict__ tq) {
  __shared__ float qrow[HD], krow[HD];
  size_t row = blockIdx.x;
  int t = threadIdx.x;
  qrow[t] = (float)qh[row * HD + t];
  krow[t] = (float)kh[row * HD + t];
  __syncthreads();
  if (t < NB) {
    float sk = 0.f, sq = 0.f;
#pragma unroll 8
    for (int d = 0; d < HD; ++d) {
      sk += qrow[d] * wk[d * NB + t];
      sq += krow[d] * wq[d * NB + t];
    }
    tk[row * NB + t] = sk;
    tq[row * NB + t] = sq * SCALE;
  }
}

// ---------------------------------------------------------------------------
// Kernel 4: fused attention block.  One workgroup per (bh, 16-row tile).
// 8 waves: each wave owns 8 key tiles for scores, then (d-tile, K-half) for PV.
// ---------------------------------------------------------------------------
__global__ __launch_bounds__(256) void attn_kernel(
    const _Float16* __restrict__ qh, const _Float16* __restrict__ kh,
    const _Float16* __restrict__ vt, const float* __restrict__ tk,
    const float* __restrict__ tq, const int* __restrict__ rp,
    const float* __restrict__ wv,                 // w_rpe_v [NB,HD] f32
    float* __restrict__ attn_out,                 // [B,H,N,N]
    _Float16* __restrict__ ctx16) {               // [B,N,C] f16
  __shared__ _Float16 P16[16][NN + 16];   // normalized probs, f16
  __shared__ float bs[16][52];            // per-row bucket sums
  __shared__ float redmax[8][16];
  __shared__ float redsum[8][16];
  __shared__ float redbuf[4][32][8];      // PV K-half reduction
  __shared__ float wvs[NB][HD];

  const int bh = blockIdx.x >> 6;
  const int i0 = (blockIdx.x & 63) << 4;
  const int b = bh / HH, h = bh % HH;
  const int tid = threadIdx.x;
  const int w = tid >> 5, lane = tid & 31;
  const int lrow = (lane & 16) ? 8 : 0;
  const int lcol = lane & 15;

  // async-stage the rpe_v table: overlaps score phase and local row max
  for (int t = tid; t < NB * HD; t += 256)
    async_f32_to_lds(wv + t, &wvs[0][0] + t);
  for (int t = tid; t < 16 * 52; t += 256) ((float*)bs)[t] = 0.f;

  // ---- scores: S = scale*(Q Kt) + gather(tk) + gather(tq)^T -------------
  const _Float16* qbase = qh + ((size_t)bh * NN + i0) * HD;
  v16h a0 = load_a_frag(qbase, HD, lane);
  v16h a1 = load_a_frag(qbase + 32, HD, lane);

  float vals[8][8];
#pragma unroll
  for (int c = 0; c < 8; ++c) {
    const int j0 = (w * 8 + c) << 4;
    const _Float16* kbase = kh + ((size_t)bh * NN + j0) * HD;
    v16h b0 = load_b_frag(kbase, HD, lane);
    v16h b1 = load_b_frag(kbase + 32, HD, lane);
    v8f acc = {};
    acc = wmma_f16(a0, b0, acc);
    acc = wmma_f16(a1, b1, acc);
    const int j = j0 + lcol;
    // transposed bucket row: 8 consecutive ints -> two b128 loads
    const int4 rt0 = *(const int4*)&rp[j * NN + i0 + lrow];
    const int4 rt1 = *(const int4*)&rp[j * NN + i0 + lrow + 4];
    const int bji[8] = {rt0.x, rt0.y, rt0.z, rt0.w, rt1.x, rt1.y, rt1.z, rt1.w};
    const float* tqj = tq + ((size_t)bh * NN + j) * NB;
#pragma unroll
    for (int r = 0; r < 8; ++r) {
      const int i = i0 + r + lrow;
      const int bij = rp[i * NN + j];
      vals[c][r] = SCALE * acc[r]
                 + tk[((size_t)bh * NN + i) * NB + bij]
                 + tqj[bji[r]];
    }
  }

  // ---- softmax: row max ------------------------------------------------
  float lmax[8];
#pragma unroll
  for (int r = 0; r < 8; ++r) {
    float m = vals[0][r];
#pragma unroll
    for (int c = 1; c < 8; ++c) m = fmaxf(m, vals[c][r]);
#pragma unroll
    for (int mk = 1; mk <= 8; mk <<= 1) m = fmaxf(m, __shfl_xor(m, mk, 32));
    lmax[r] = m;
  }
  wait_asynccnt0();                      // wvs staged before first barrier
  if (lcol == 0) {
#pragma unroll
    for (int r = 0; r < 8; ++r) redmax[w][r + lrow] = lmax[r];
  }
  __syncthreads();
  float gmax[8];
#pragma unroll
  for (int r = 0; r < 8; ++r) {
    float m = -3.0e38f;
#pragma unroll
    for (int ww = 0; ww < 8; ++ww) m = fmaxf(m, redmax[ww][r + lrow]);
    gmax[r] = m;
  }
  // ---- exp + row sum ---------------------------------------------------
  float lsum[8];
#pragma unroll
  for (int r = 0; r < 8; ++r) lsum[r] = 0.f;
#pragma unroll
  for (int c = 0; c < 8; ++c)
#pragma unroll
    for (int r = 0; r < 8; ++r) {
      vals[c][r] = __expf(vals[c][r] - gmax[r]);
      lsum[r] += vals[c][r];
    }
#pragma unroll
  for (int r = 0; r < 8; ++r)
#pragma unroll
    for (int mk = 1; mk <= 8; mk <<= 1) lsum[r] += __shfl_xor(lsum[r], mk, 32);
  if (lcol == 0) {
#pragma unroll
    for (int r = 0; r < 8; ++r) redsum[w][r + lrow] = lsum[r];
  }
  __syncthreads();
  float inv[8];
#pragma unroll
  for (int r = 0; r < 8; ++r) {
    float s = 0.f;
#pragma unroll
    for (int ww = 0; ww < 8; ++ww) s += redsum[ww][r + lrow];
    inv[r] = 1.0f / s;
  }

  // ---- write attn (f32, NT stores) + P16 (f16, LDS) --------------------
  float* aout = attn_out + ((size_t)bh * NN + i0) * NN;
#pragma unroll
  for (int c = 0; c < 8; ++c) {
    const int j = ((w * 8 + c) << 4) + lcol;
#pragma unroll
    for (int r = 0; r < 8; ++r) {
      const float p = vals[c][r] * inv[r];
      __builtin_nontemporal_store(p, &aout[(size_t)(r + lrow) * NN + j]);
      P16[r + lrow][j] = (_Float16)p;
    }
  }
  __syncthreads();

  // ---- per-row bucket segment sums (normalized probs) ------------------
  for (int t = tid; t < 16 * NN; t += 256) {
    const int rr = t >> 10, j = t & (NN - 1);
    const float p = (float)P16[rr][j];
    const int bkt = rp[(i0 + rr) * NN + j];
    atomicAdd(&bs[rr][bkt], p);
  }
  __syncthreads();

  // ---- out = P @ V  (+ bucket_sums @ w_rpe_v) --------------------------
  const int d0 = (w & 3) << 4;
  const int kh0 = (w >> 2) << 9;       // 0 or 512
  const _Float16* vbase = vt + ((size_t)bh * HD + d0) * NN + kh0;
  v8f acc = {};
#pragma unroll
  for (int t = 0; t < 16; ++t) {
    const int kk = t * 32;
    v16h a = load_a_frag(&P16[0][kh0 + kk], NN + 16, lane);
    v16h bf = load_b_frag(vbase + kk, NN, lane);
    acc = wmma_f16(a, bf, acc);
  }
  if (w >= 4) {
#pragma unroll
    for (int r = 0; r < 8; ++r) redbuf[w - 4][lane][r] = acc[r];
  }
  __syncthreads();
  if (w < 4) {
    const int d = d0 + lcol;
#pragma unroll
    for (int r = 0; r < 8; ++r) {
      const int row = r + lrow;
      float s = acc[r] + redbuf[w][lane][r];
#pragma unroll 7
      for (int u = 0; u < NB; ++u) s += bs[row][u] * wvs[u][d];
      const int n = i0 + row;
      ctx16[((size_t)b * NN + n) * CC + h * HD + d] = (_Float16)s;
    }
  }
}

// ---------------------------------------------------------------------------
// Kernel 5: out = ctx @ proj_w^T + proj_b   ([B*N,768] x [768,768])
// ---------------------------------------------------------------------------
__global__ __launch_bounds__(256) void proj_kernel(
    const _Float16* __restrict__ ctx16, const _Float16* __restrict__ w16,
    const float* __restrict__ proj_b, float* __restrict__ out) {
  __shared__ _Float16 As[16][CC + 16];
  const int m0 = blockIdx.x << 4;
  for (int t = threadIdx.x; t < 16 * (CC / 8); t += 256) {
    const int r = t / (CC / 8), ch = t % (CC / 8);
    *(v8h*)&As[r][ch * 8] = *(const v8h*)&ctx16[((size_t)(m0 + r)) * CC + ch * 8];
  }
  __syncthreads();
  const int w = threadIdx.x >> 5, lane = threadIdx.x & 31;
  const int lrow = (lane & 16) ? 8 : 0, lcol = lane & 15;
  const int co0 = blockIdx.y * 128 + w * 16;
  v8f acc = {};
#pragma unroll
  for (int k0 = 0; k0 < CC; k0 += 32) {
    v16h a = load_a_frag(&As[0][k0], CC + 16, lane);
    v16h bf = load_b_frag(w16 + (size_t)co0 * CC + k0, CC, lane);
    acc = wmma_f16(a, bf, acc);
  }
  const int col = co0 + lcol;
  const float bias = proj_b[col];
#pragma unroll
  for (int r = 0; r < 8; ++r)
    __builtin_nontemporal_store(acc[r] + bias,
                                &out[(size_t)(m0 + r + lrow) * CC + col]);
}

// ---------------------------------------------------------------------------
extern "C" void kernel_launch(void* const* d_in, const int* in_sizes, int n_in,
                              void* d_out, int out_size, void* d_ws, size_t ws_size,
                              hipStream_t stream) {
  (void)in_sizes; (void)n_in; (void)out_size; (void)ws_size;
  const float* q      = (const float*)d_in[0];
  const float* k      = (const float*)d_in[1];
  const float* v      = (const float*)d_in[2];
  const float* w_rpeq = (const float*)d_in[3];
  const float* w_rpek = (const float*)d_in[4];
  const float* w_rpev = (const float*)d_in[5];
  const float* proj_w = (const float*)d_in[6];
  const float* proj_b = (const float*)d_in[7];
  const int*   rp     = (const int*)d_in[8];

  float* out  = (float*)d_out;
  float* attn = out + (size_t)BB * NN * CC;

  // workspace carve-up (256B aligned); total ~90.2 MB
  char* ws = (char*)d_ws;
  size_t off = 0;
  auto carve = [&](size_t bytes) {
    char* p = ws + off;
    off = (off + bytes + 255) & ~(size_t)255;
    return p;
  };
  const size_t bhnd = (size_t)BB * HH * NN * HD;
  _Float16* qh16  = (_Float16*)carve(bhnd * 2);
  _Float16* kh16  = (_Float16*)carve(bhnd * 2);
  _Float16* vt16  = (_Float16*)carve(bhnd * 2);
  float*    tk    = (float*)carve((size_t)BB * HH * NN * NB * 4);
  float*    tq    = (float*)carve((size_t)BB * HH * NN * NB * 4);
  _Float16* ctx16 = (_Float16*)carve((size_t)BB * NN * CC * 2);
  _Float16* w16   = (_Float16*)carve((size_t)CC * CC * 2);

  const size_t tot = (size_t)BB * NN * CC;
  pack_qkv_kernel<<<(unsigned)((tot + 255) / 256), 256, 0, stream>>>(
      q, k, v, qh16, kh16, vt16);
  pack_w_kernel<<<(CC * CC + 255) / 256, 256, 0, stream>>>(proj_w, w16);
  bias_kernel<<<BB * HH * NN, HD, 0, stream>>>(qh16, kh16, w_rpek, w_rpeq, tk, tq);
  attn_kernel<<<BB * HH * (NN / 16), 256, 0, stream>>>(
      qh16, kh16, vt16, tk, tq, rp, w_rpev, attn, ctx16);
  proj_kernel<<<dim3((BB * NN) / 16, CC / 128), 256, 0, stream>>>(
      ctx16, w16, proj_b, out);
}
